// HyenaForQuestionAnswering_50208167690721
// MI455X (gfx1250) — matrix operand order repaired and checked by reading, not compiled, not run on hardware
//
#include <hip/hip_runtime.h>
#include <hip/hip_bf16.h>
#include <stdint.h>

// ---------------------------------------------------------------------------
// Hyena QA head for MI455X (gfx1250, wave32, WMMA + async-to-LDS).
//
// Depthwise long conv (K=4095) as block-Toeplitz GEMM on v_wmma_f32_16x16x32_bf16:
//   C[i][n] = sum_sc  A_sc[i][s] * B_sc[s][n]
//   A_sc[i][s] = w[32*sc + s - 16*i]          (zero-padded filter, per channel)
//   B_sc[s][n] = xp[t0 + n + 32*sc + s]       (zero-padded signal)
//   => C[i][n] = y[t0 + 16*i + n]             (256 consecutive outputs / tile)
// 136 chunks (sc) cover k in [0,4095) for every i in [0,16).
//
// B is a Hankel window whose per-lane base parity == (n & 1), a lane constant.
// We keep the signal twice in LDS (original + shifted-by-one copy) so every
// lane does aligned dword loads with zero shift VALU work per WMMA.
// ---------------------------------------------------------------------------

typedef __attribute__((ext_vector_type(16))) __bf16 v16bf;
typedef __attribute__((ext_vector_type(8)))  float  v8f;

constexpr int Dm = 768, NLay = 12, Kf = 4095, Bb = 2, Ls = 4095;
constexpr int Pp   = 2047;   // K//2 (SAME padding)
constexpr int XS   = 8448;   // padded signal stride (>= L+2P=8189, >= max read 8207)
constexpr int WLEN = 4608;   // 256 leading zeros + 4095 taps + zero tail
constexpr int NCH  = 136;    // s-chunks of 32: covers 32*sc+s in [0,4352)

// LDS layout (bytes) inside one combined array (deterministic offset 0):
//   xs  : [0,           2*XS)        original signal + 16B zero pad
//   xs1 : [XS1_OFF,     XS1_OFF+2*XS) signal shifted by one element
//   wsh : [WSH_OFF,     WSH_OFF+2*WLEN)
constexpr int XS_PAD   = 8;                 // shorts of zero pad after xs
constexpr int XS1_OFFS = XS + XS_PAD;       // short index of xs1
constexpr int WSH_OFFS = XS1_OFFS + XS;     // short index of wsh
constexpr int SMEM_SH  = WSH_OFFS + WLEN;   // total shorts (43,024 B)

__device__ __forceinline__ unsigned short f2bf(float f) {
  union { float f; uint32_t u; } x; x.f = f;
  uint32_t u = x.u;
  uint32_t r = u + 0x7FFFu + ((u >> 16) & 1u);   // round-to-nearest-even
  return (unsigned short)(r >> 16);
}

// Async global->LDS 16B copy (gfx1250 TDM-lite path, tracked by ASYNCcnt).
__device__ __forceinline__ void async_copy16(unsigned lds_byte_off, const void* gptr) {
  asm volatile("global_load_async_to_lds_b128 %0, %1, off"
               :: "v"(lds_byte_off), "v"(gptr) : "memory");
}
__device__ __forceinline__ void wait_async0() {
  asm volatile("s_wait_asynccnt 0x0" ::: "memory");
}

// ---------------------------------------------------------------------------
__global__ __launch_bounds__(256)
void zero16_kernel(uint4* __restrict__ p, size_t n16) {
  size_t i = (size_t)blockIdx.x * 256 + threadIdx.x;
  if (i < n16) p[i] = make_uint4(0u, 0u, 0u, 0u);
}

// filt_w (NL,D,K) fp32 -> per-(layer,channel) zero-padded bf16 filter of WLEN
__global__ __launch_bounds__(256)
void wprep_kernel(const float* __restrict__ fw, unsigned short* __restrict__ wpad,
                  size_t total) {
  size_t i = (size_t)blockIdx.x * 256 + threadIdx.x;
  if (i >= total) return;
  int    j  = (int)(i % WLEN);
  size_t ld = i / WLEN;                       // l*D + d
  int    k  = j - 256;
  float  v  = (k >= 0 && k < Kf) ? fw[ld * (size_t)Kf + k] : 0.0f;
  wpad[ld * (size_t)WLEN + j] = f2bf(v);
}

// x = emb[ids] + pos  -> channel-major fp32 [b][d][t] and bf16 padded [b][d][P+t]
__global__ __launch_bounds__(256)
void embed_kernel(const int* __restrict__ ids, const float* __restrict__ emb,
                  const float* __restrict__ pos, float* __restrict__ xf32,
                  unsigned short* __restrict__ xpad, size_t total) {
  size_t i = (size_t)blockIdx.x * 256 + threadIdx.x;
  if (i >= total) return;
  int    t = (int)(i % Ls);
  size_t r = i / Ls;
  int    d = (int)(r % Dm);
  int    b = (int)(r / Dm);
  int    id = ids[(size_t)b * Ls + t];
  float  v  = emb[(size_t)id * Dm + d] + pos[(size_t)t * Dm + d];
  size_t bd = (size_t)b * Dm + d;
  xf32[bd * (size_t)Ls + t]      = v;
  xpad[bd * (size_t)XS + Pp + t] = f2bf(v);
}

// ---------------------------------------------------------------------------
// One residual long-conv layer. Block = 4 waves = one (b,d) signal.
// Wave w accumulates tiles t0 = (4w+j)*256, j=0..3 (A-fragment reused 4x).
// ---------------------------------------------------------------------------
__global__ __launch_bounds__(128)
void hyena_conv_layer(const unsigned short* __restrict__ xpad_in,
                      const float* __restrict__ xf32_in,
                      const unsigned short* __restrict__ wpad_l,  // + l*D*WLEN
                      const float* __restrict__ fb_l,             // + l*D
                      unsigned short* __restrict__ xpad_out,
                      float* __restrict__ xf32_out) {
  __shared__ __align__(16) unsigned short smem[SMEM_SH];
  unsigned short* xs  = smem;
  unsigned short* xs1 = smem + XS1_OFFS;
  unsigned short* wsh = smem + WSH_OFFS;

  const int bd  = blockIdx.x;           // b*D + d
  const int d   = bd % Dm;
  const int tid = threadIdx.x;

  // ---- Stage signal + filter into LDS with async b128 copies -------------
  {
    const uint4* xsrc = (const uint4*)(xpad_in + (size_t)bd * XS);
    for (int i = tid; i < XS / 8; i += 128)
      async_copy16((unsigned)(i * 16), xsrc + i);
    const uint4* wsrc = (const uint4*)(wpad_l + (size_t)d * WLEN);
    for (int i = tid; i < WLEN / 8; i += 128)
      async_copy16((unsigned)(WSH_OFFS * 2 + i * 16), wsrc + i);
    if (tid < 4) ((uint32_t*)smem)[XS / 2 + tid] = 0u;   // zero pad after xs
  }
  wait_async0();
  __syncthreads();

  // ---- Build shifted copy: xs1[j] = xs[j+1] (one-time funnel pass) -------
  {
    const uint32_t* s32 = (const uint32_t*)xs;
    uint32_t*       d32 = (uint32_t*)xs1;
    for (int i = tid; i < XS / 2; i += 128)
      d32[i] = (s32[i] >> 16) | (s32[i + 1] << 16);
  }
  __syncthreads();

  const int  lane = tid & 31;
  const int  wv   = tid >> 5;
  const int  n    = lane & 15;          // N column / A row M
  const bool hiL  = lane >= 16;

  // Per-lane constants: odd lanes read the shifted copy; base is then even.
  const uint32_t* bbase = (const uint32_t*)((n & 1) ? xs1 : xs);
  const int bcol = (((hiL ? 16 : 0) + (n & ~1)) >> 1);
  const int t0base = wv * 1024;

  union AV { uint4    q[2]; v16bf v; };
  union BV { uint32_t w[8]; v16bf v; };

  v8f acc[4] = {v8f{}, v8f{}, v8f{}, v8f{}};

  for (int sc = 0; sc < NCH; ++sc) {
    // A-fragment (contiguous k = 16h+s order, same as B): lane m<16 holds K={0..15},
    // lanes 16..31 hold K={16..31}; value = wsh[256 + 32sc + K - 16M].
    // Both 8-tap runs are 16B-aligned -> two ds_load_b128.
    AV a;
    const int abase = 256 + 32 * sc + (hiL ? 16 : 0) - 16 * n;
    a.q[0] = *(const uint4*)(wsh + abase);
    a.q[1] = *(const uint4*)(wsh + abase + 8);

    const int bidx0 = ((t0base + 32 * sc) >> 1) + bcol;
#pragma unroll
    for (int j = 0; j < 4; ++j) {
      // B-fragment (contiguous k order, same as A): dword e = bf16 pair at rows K={2e,2e+1}
      // (+16 for high lanes), column n: 8 aligned dword loads, one base VGPR.
      const uint32_t* p = bbase + (bidx0 + j * 128);
      BV b;
#pragma unroll
      for (int e = 0; e < 8; ++e) b.w[e] = p[e];
      acc[j] = __builtin_amdgcn_wmma_f32_16x16x32_bf16(
          false, a.v, false, b.v, (short)0, acc[j], false, false);
    }
  }

  // Store: C element e of lane -> y[t0 + 16*(e + hi*8) + n]; add bias+residual,
  // emit fp32 for next residual and bf16 into the next layer's padded buffer.
  const float  bias = fb_l[d];
  const size_t rowL = (size_t)bd * Ls;
  const size_t rowX = (size_t)bd * XS;
#pragma unroll
  for (int j = 0; j < 4; ++j) {
    const int t0 = t0base + j * 256;
#pragma unroll
    for (int e = 0; e < 8; ++e) {
      const int M = e + (hiL ? 8 : 0);
      const int t = t0 + 16 * M + n;
      if (t < Ls) {
        float y = acc[j][e] + bias + xf32_in[rowL + t];
        xf32_out[rowL + t]      = y;
        xpad_out[rowX + Pp + t] = f2bf(y);
      }
    }
  }
}

// ---------------------------------------------------------------------------
// LayerNorm over D + (D x 2) projection. One thread per (b,t); reads are
// coalesced across t for each fixed d.
__global__ __launch_bounds__(256)
void lnproj_kernel(const float* __restrict__ x, const float* __restrict__ g,
                   const float* __restrict__ bln, const float* __restrict__ qw,
                   const float* __restrict__ qb, float* __restrict__ out) {
  int i = blockIdx.x * 256 + threadIdx.x;
  if (i >= Bb * Ls) return;
  int b = i / Ls, t = i % Ls;
  const float* xb = x + (size_t)b * Dm * Ls + t;
  float s = 0.f, s2 = 0.f;
  for (int d = 0; d < Dm; ++d) {
    float v = xb[(size_t)d * Ls];
    s += v; s2 += v * v;
  }
  float m    = s * (1.0f / Dm);
  float var  = s2 * (1.0f / Dm) - m * m;
  float rinv = rsqrtf(var + 1e-5f);
  float l0 = qb[0], l1 = qb[1];
  for (int d = 0; d < Dm; ++d) {
    float v = (xb[(size_t)d * Ls] - m) * rinv * g[d] + bln[d];
    l0 += v * qw[d * 2 + 0];
    l1 += v * qw[d * 2 + 1];
  }
  out[i]                    = l0;   // start logits
  out[(size_t)Bb * Ls + i]  = l1;   // end logits
}

// ---------------------------------------------------------------------------
extern "C" void kernel_launch(void* const* d_in, const int* in_sizes, int n_in,
                              void* d_out, int out_size, void* d_ws, size_t ws_size,
                              hipStream_t stream) {
  const int*   ids = (const int*)d_in[0];
  const float* emb = (const float*)d_in[1];
  const float* pos = (const float*)d_in[2];
  const float* fw  = (const float*)d_in[3];
  const float* fb  = (const float*)d_in[4];
  const float* g   = (const float*)d_in[5];
  const float* bln = (const float*)d_in[6];
  const float* qw  = (const float*)d_in[7];
  const float* qb  = (const float*)d_in[8];
  float*       out = (float*)d_out;

  char* ws = (char*)d_ws;
  const size_t sz_wpad = (size_t)NLay * Dm * WLEN * 2;   //  85.0 MB
  const size_t sz_xpad = (size_t)Bb * Dm * XS * 2;       //  26.0 MB (x2)
  const size_t sz_xf32 = (size_t)Bb * Dm * Ls * 4;       //  25.2 MB (x2)
  unsigned short* wpad   = (unsigned short*)ws;
  unsigned short* xpad_a = (unsigned short*)(ws + sz_wpad);
  unsigned short* xpad_b = (unsigned short*)(ws + sz_wpad + sz_xpad);
  float*          xf32_a = (float*)(ws + sz_wpad + 2 * sz_xpad);
  float*          xf32_b = (float*)(ws + sz_wpad + 2 * sz_xpad + sz_xf32);

  // Zero both padded bf16 buffers (halo regions must read as 0 every call).
  {
    size_t n16 = (2 * sz_xpad) / 16;
    zero16_kernel<<<(unsigned)((n16 + 255) / 256), 256, 0, stream>>>((uint4*)xpad_a, n16);
  }
  // Filters -> padded bf16 (done per call; deterministic).
  {
    size_t nw = (size_t)NLay * Dm * WLEN;
    wprep_kernel<<<(unsigned)((nw + 255) / 256), 256, 0, stream>>>(fw, wpad, nw);
  }
  // Embedding + positional -> channel-major activations.
  {
    size_t ne = (size_t)Bb * Ls * Dm;
    embed_kernel<<<(unsigned)((ne + 255) / 256), 256, 0, stream>>>(ids, emb, pos,
                                                                   xf32_a, xpad_a, ne);
  }
  // 12 residual long-conv layers (ping-pong).
  unsigned short* xi = xpad_a; float* fi = xf32_a;
  unsigned short* xo = xpad_b; float* fo = xf32_b;
  for (int l = 0; l < NLay; ++l) {
    hyena_conv_layer<<<Bb * Dm, 128, 0, stream>>>(
        xi, fi, wpad + (size_t)l * Dm * WLEN, fb + (size_t)l * Dm, xo, fo);
    unsigned short* tx = xi; xi = xo; xo = tx;
    float* tf = fi; fi = fo; fo = tf;
  }
  // LayerNorm + QA projection.
  lnproj_kernel<<<(Bb * Ls + 255) / 256, 256, 0, stream>>>(fi, g, bln, qw, qb, out);
  (void)in_sizes; (void)n_in; (void)out_size; (void)ws_size;
}
